// MultiHead_Attention_40278203302170
// MI455X (gfx1250) — compile-verified, matrix operand order checked by reading, not run on hardware
//
#include <hip/hip_runtime.h>
#include <hip/hip_bf16.h>

typedef __attribute__((ext_vector_type(16))) _Float16 v16h;
typedef __attribute__((ext_vector_type(8)))  float    v8f;

// Problem constants
#define LQ   2048
#define BB   4
#define DD_  1024
#define HH   16
#define HD   64

// K index inside a 16-bit A/B fragment VGPR v (pairs), per CDNA5 ISA 7.12.2:
// lanes 0-15: v0..3 -> K{0..7}, v4..7 -> K{16..23}; lanes 16-31: +8
__device__ __forceinline__ int kb16(int v, int half) {
    return ((v < 4) ? (2 * v) : (8 + 2 * v)) + half * 8;
}

union Frag16 { v16h v; unsigned u[8]; _Float16 h[16]; };

static __device__ __forceinline__ v8f wmma_f16(v16h a, v16h b, v8f c) {
    return __builtin_amdgcn_wmma_f32_16x16x32_f16(false, a, false, b, (short)0, c, false, false);
}

// ---------------------------------------------------------------------------
// f32 -> f16 conversion
// ---------------------------------------------------------------------------
__global__ void cvt_f32_to_f16(const float* __restrict__ src, _Float16* __restrict__ dst, int n) {
    int i = blockIdx.x * 256 + threadIdx.x;
    if (i < n) dst[i] = (_Float16)src[i];
}

// ---------------------------------------------------------------------------
// GEMM: C(8192x1024) = A(8192x1024,f16) * W^T(1024x1024,f16) + bias
// One block per 16-row M tile (512 blocks x 512 threads = 16 waves).
// The block's shared 16x1024 A panel (32 KB) is staged in LDS with
// global_load_async_to_lds_b128 (ASYNCcnt path), then each wave computes a
// 16x64 output tile, reading A fragments via ds_load_b128 and software-
// pipelining its B-fragment global loads one k-step ahead.
// head_split=1: write f16 to head layout [b*H+h][l][d] scaled; else f32.
// ---------------------------------------------------------------------------
__global__ void __launch_bounds__(512)
gemm_wmma(const _Float16* __restrict__ A, const _Float16* __restrict__ W,
          const float* __restrict__ bias, _Float16* __restrict__ dstH,
          float* __restrict__ dstF, float scale, int head_split)
{
    __shared__ __align__(16) _Float16 alds[16][DD_];   // 32 KB A panel

    int lane = threadIdx.x & 31;
    int wave = threadIdx.x >> 5;      // 16 waves = 16 N blocks of 64
    int mt   = blockIdx.x;            // 512 M tiles
    int n64  = wave;
    int half = lane >> 4;
    int lm   = lane & 15;

    // ---- async stage of the A panel: 2048 x 16B chunks, 4 per thread ----
#pragma unroll
    for (int i = 0; i < 4; ++i) {
        int c    = threadIdx.x + i * 512;       // 0..2047
        int row  = c >> 7;                      // 0..15
        int colh = (c & 127) * 8;               // halves
        unsigned ldsoff = (unsigned)(uintptr_t)&alds[row][colh];
        unsigned long long ga =
            (unsigned long long)(A + (size_t)(mt * 16 + row) * DD_ + colh);
        asm volatile("global_load_async_to_lds_b128 %0, %1, off"
                     :: "v"(ldsoff), "v"(ga) : "memory");
    }
    asm volatile("s_wait_asynccnt 0x0" ::: "memory");
    __syncthreads();

    const _Float16* wrow = W + (size_t)(n64 * 64 + lm) * DD_;

    v8f z = {0,0,0,0,0,0,0,0};
    v8f acc[4];
    acc[0] = z; acc[1] = z; acc[2] = z; acc[3] = z;

    // prologue: B fragments for k0 = 0
    Frag16 bcur[4];
#pragma unroll
    for (int s = 0; s < 4; ++s)
#pragma unroll
        for (int v = 0; v < 8; ++v)
            bcur[s].u[v] = *(const unsigned*)(wrow + (size_t)s * 16 * DD_ + kb16(v, half));

    for (int k0 = 0; k0 < DD_; k0 += 32) {
        // A fragment from LDS: two 16B ds loads
        Frag16 a;
        *(uint4*)&a.u[0] = *(const uint4*)&alds[lm][k0 + half * 8];
        *(uint4*)&a.u[4] = *(const uint4*)&alds[lm][k0 + 16 + half * 8];

        // prefetch next k-step's B fragments (pipelined past the WMMAs)
        Frag16 bnxt[4];
        int k1 = k0 + 32;
        if (k1 < DD_) {
#pragma unroll
            for (int s = 0; s < 4; ++s)
#pragma unroll
                for (int v = 0; v < 8; ++v)
                    bnxt[s].u[v] =
                        *(const unsigned*)(wrow + (size_t)s * 16 * DD_ + k1 + kb16(v, half));
        }

#pragma unroll
        for (int s = 0; s < 4; ++s)
            acc[s] = wmma_f16(a.v, bcur[s].v, acc[s]);

        if (k1 < DD_) {
#pragma unroll
            for (int s = 0; s < 4; ++s)
                bcur[s] = bnxt[s];
        }
    }

#pragma unroll
    for (int s = 0; s < 4; ++s) {
        int nout = n64 * 64 + s * 16 + lm;
        float bv = bias[nout];
#pragma unroll
        for (int r = 0; r < 8; ++r) {
            int mout = mt * 16 + half * 8 + r;
            float val = (acc[s][r] + bv) * scale;
            if (head_split) {
                int l = mout >> 2, b = mout & 3;           // m = l*B + b
                int h = nout >> 6, d = nout & 63;          // n = h*64 + d
                dstH[(((size_t)(b * HH + h) * LQ + l) << 6) + d] = (_Float16)val;
            } else {
                dstF[(size_t)mout * DD_ + nout] = val;
            }
        }
    }
}

// ---------------------------------------------------------------------------
// Pass 1: per-row softmax stats (running max m, running sum l) per (n, q-row)
// One wave per (n, q-tile-of-16). Score rows live in one C-fragment VGPR,
// spread across 16 lanes of a half-wave -> xor-shuffle reductions.
// ---------------------------------------------------------------------------
__global__ void __launch_bounds__(256)
stats_kernel(const _Float16* __restrict__ Qh, const _Float16* __restrict__ Kh,
             float* __restrict__ mstat, float* __restrict__ lstat)
{
    int lane = threadIdx.x & 31;
    int wave = threadIdx.x >> 5;
    int task = blockIdx.x * 8 + wave;   // 8192 = 64 n * 128 qt
    int n  = task >> 7;
    int qt = task & 127;
    int half = lane >> 4, lm = lane & 15;

    const _Float16* qrow = Qh + ((size_t)n * LQ + qt * 16 + lm) * HD;
    Frag16 a0, a1;
#pragma unroll
    for (int v = 0; v < 8; ++v) {
        int kb = kb16(v, half);
        a0.u[v] = *(const unsigned*)(qrow + kb);
        a1.u[v] = *(const unsigned*)(qrow + 32 + kb);
    }

    float mrun[8], lrun[8];
#pragma unroll
    for (int r = 0; r < 8; ++r) { mrun[r] = -1e30f; lrun[r] = 0.f; }

    for (int kt = 0; kt < LQ / 16; ++kt) {
        const _Float16* kr = Kh + ((size_t)n * LQ + kt * 16 + lm) * HD;
        Frag16 b0, b1;
#pragma unroll
        for (int v = 0; v < 8; ++v) {
            int kb = kb16(v, half);
            b0.u[v] = *(const unsigned*)(kr + kb);
            b1.u[v] = *(const unsigned*)(kr + 32 + kb);
        }
        v8f s = {0,0,0,0,0,0,0,0};
        s = wmma_f16(a0.v, b0.v, s);
        s = wmma_f16(a1.v, b1.v, s);

#pragma unroll
        for (int r = 0; r < 8; ++r) {
            float x  = s[r];
            float tm = x;
#pragma unroll
            for (int off = 1; off <= 8; off <<= 1)
                tm = fmaxf(tm, __shfl_xor(tm, off, 32));
            float mnew = fmaxf(mrun[r], tm);
            float e  = __expf(x - mnew);
            float ts = e;
#pragma unroll
            for (int off = 1; off <= 8; off <<= 1)
                ts += __shfl_xor(ts, off, 32);
            lrun[r] = lrun[r] * __expf(mrun[r] - mnew) + ts;
            mrun[r] = mnew;
        }
    }

    if (lm == 0) {
#pragma unroll
        for (int r = 0; r < 8; ++r) {
            int row = qt * 16 + half * 8 + r;
            mstat[(size_t)n * LQ + row] = mrun[r];
            lstat[(size_t)n * LQ + row] = lrun[r];
        }
    }
}

// ---------------------------------------------------------------------------
// Pass 2: one block per (b, q-tile), 16 waves = 16 heads.
//  - recompute scores via WMMA, normalize with precomputed stats
//  - stage P (f16) in per-wave LDS region: doubles as transpose for P*V and
//    as the head-reduction buffer for avg_weights (no global atomics)
//  - accumulate P*V into per-wave f32 accumulators, write merged f16 attn
// ---------------------------------------------------------------------------
__global__ void __launch_bounds__(512)
attn_kernel(const _Float16* __restrict__ Qh, const _Float16* __restrict__ Kh,
            const _Float16* __restrict__ Vh,
            const float* __restrict__ mstat, const float* __restrict__ lstat,
            _Float16* __restrict__ attn16, float* __restrict__ avg)
{
    __shared__ __align__(16) _Float16 plds[16][16 * 32];  // 16 KB: per-head 16x32 P tile

    int lane = threadIdx.x & 31;
    int wave = threadIdx.x >> 5;      // head
    int qt = blockIdx.x;              // 128
    int b  = blockIdx.y;              // 4
    int n  = b * HH + wave;
    int half = lane >> 4, lm = lane & 15;

    const _Float16* qrow = Qh + ((size_t)n * LQ + qt * 16 + lm) * HD;
    Frag16 a0, a1;
#pragma unroll
    for (int v = 0; v < 8; ++v) {
        int kb = kb16(v, half);
        a0.u[v] = *(const unsigned*)(qrow + kb);
        a1.u[v] = *(const unsigned*)(qrow + 32 + kb);
    }

    float mr[8], li[8];
#pragma unroll
    for (int r = 0; r < 8; ++r) {
        int row = qt * 16 + half * 8 + r;
        mr[r] = mstat[(size_t)n * LQ + row];
        li[r] = 1.0f / lstat[(size_t)n * LQ + row];
    }

    v8f z = {0,0,0,0,0,0,0,0};
    v8f acc[4];
    acc[0] = z; acc[1] = z; acc[2] = z; acc[3] = z;

    for (int kc = 0; kc < LQ / 32; ++kc) {
#pragma unroll
        for (int t = 0; t < 2; ++t) {
            int kt = kc * 2 + t;
            const _Float16* kr = Kh + ((size_t)n * LQ + kt * 16 + lm) * HD;
            Frag16 b0, b1;
#pragma unroll
            for (int v = 0; v < 8; ++v) {
                int kb = kb16(v, half);
                b0.u[v] = *(const unsigned*)(kr + kb);
                b1.u[v] = *(const unsigned*)(kr + 32 + kb);
            }
            v8f s = z;
            s = wmma_f16(a0.v, b0.v, s);
            s = wmma_f16(a1.v, b1.v, s);
#pragma unroll
            for (int r = 0; r < 8; ++r) {
                float p = __expf(s[r] - mr[r]) * li[r];
                plds[wave][(half * 8 + r) * 32 + t * 16 + lm] = (_Float16)p;
            }
        }
        __syncthreads();

        // avg_weights: sum 16 head tiles, one element per thread (512 = 16x32)
        {
            int idx = threadIdx.x;
            int row = idx >> 5, col = idx & 31;
            float sum = 0.f;
#pragma unroll
            for (int w = 0; w < 16; ++w)
                sum += (float)plds[w][row * 32 + col];
            avg[(size_t)b * LQ * LQ + (size_t)(qt * 16 + row) * LQ + kc * 32 + col] =
                sum * (1.0f / 16.0f);
        }

        // P*V: read own P tile back in A-fragment order (LDS transpose)
        Frag16 pa;
#pragma unroll
        for (int v = 0; v < 8; ++v)
            pa.u[v] = *(const unsigned*)&plds[wave][lm * 32 + kb16(v, half)];

#pragma unroll
        for (int dt = 0; dt < 4; ++dt) {
            int dcol = dt * 16 + lm;
            const _Float16* vb = Vh + ((size_t)n * LQ + kc * 32) * HD + dcol;
            Frag16 bv;
#pragma unroll
            for (int v = 0; v < 8; ++v) {
                int kb = kb16(v, half);
                bv.h[2 * v]     = vb[(size_t)kb * HD];
                bv.h[2 * v + 1] = vb[(size_t)(kb + 1) * HD];
            }
            acc[dt] = wmma_f16(pa.v, bv.v, acc[dt]);
        }
        __syncthreads();
    }

    // write merged (L,B,D) f16 attention output for the final projection
#pragma unroll
    for (int dt = 0; dt < 4; ++dt) {
#pragma unroll
        for (int r = 0; r < 8; ++r) {
            int q = qt * 16 + half * 8 + r;
            int m = q * BB + b;
            int col = wave * HD + dt * 16 + lm;
            attn16[(size_t)m * DD_ + col] = (_Float16)acc[dt][r];
        }
    }
}

// ---------------------------------------------------------------------------
// Workspace layout (element offsets in f16 halves, then f32 stats)
// ---------------------------------------------------------------------------
static constexpr size_t E_LBD = (size_t)LQ * BB * DD_;   // 8,388,608
static constexpr size_t E_DD  = (size_t)DD_ * DD_;       // 1,048,576
static constexpr size_t OFF_XQ = 0;
static constexpr size_t OFF_XK = OFF_XQ + E_LBD;
static constexpr size_t OFF_XV = OFF_XK + E_LBD;
static constexpr size_t OFF_WQ = OFF_XV + E_LBD;
static constexpr size_t OFF_WK = OFF_WQ + E_DD;
static constexpr size_t OFF_WV = OFF_WK + E_DD;
static constexpr size_t OFF_WO = OFF_WV + E_DD;
static constexpr size_t OFF_QH = OFF_WO + E_DD;
static constexpr size_t OFF_KH = OFF_QH + E_LBD;
static constexpr size_t OFF_VH = OFF_KH + E_LBD;
static constexpr size_t OFF_AT = OFF_VH + E_LBD;
static constexpr size_t OFF_END_H = OFF_AT + E_LBD;
static constexpr size_t STATS_BYTES_OFF = OFF_END_H * 2;
static constexpr size_t WS_NEEDED = STATS_BYTES_OFF + 2ull * 64 * LQ * 4;

extern "C" void kernel_launch(void* const* d_in, const int* in_sizes, int n_in,
                              void* d_out, int out_size, void* d_ws, size_t ws_size,
                              hipStream_t stream)
{
    (void)in_sizes; (void)n_in; (void)out_size;
    if (ws_size < WS_NEEDED) return;

    const float* query = (const float*)d_in[0];
    const float* key_  = (const float*)d_in[1];
    const float* value = (const float*)d_in[2];
    const float* Wq = (const float*)d_in[3];
    const float* bq = (const float*)d_in[4];
    const float* Wk = (const float*)d_in[5];
    const float* bk = (const float*)d_in[6];
    const float* Wv = (const float*)d_in[7];
    const float* bv = (const float*)d_in[8];
    const float* Wo = (const float*)d_in[9];
    const float* bo = (const float*)d_in[10];

    _Float16* hb = (_Float16*)d_ws;
    _Float16* xq16 = hb + OFF_XQ;
    _Float16* xk16 = hb + OFF_XK;
    _Float16* xv16 = hb + OFF_XV;
    _Float16* wq16 = hb + OFF_WQ;
    _Float16* wk16 = hb + OFF_WK;
    _Float16* wv16 = hb + OFF_WV;
    _Float16* wo16 = hb + OFF_WO;
    _Float16* qh   = hb + OFF_QH;
    _Float16* kh   = hb + OFF_KH;
    _Float16* vh   = hb + OFF_VH;
    _Float16* at16 = hb + OFF_AT;
    float* mstat = (float*)((char*)d_ws + STATS_BYTES_OFF);
    float* lstat = mstat + (size_t)64 * LQ;

    float* out = (float*)d_out;
    float* avg = out + E_LBD;   // (B, L, L) after (L, B, D)

    const int nLBD = (int)E_LBD, nDD = (int)E_DD;
    cvt_f32_to_f16<<<(nLBD + 255) / 256, 256, 0, stream>>>(query, xq16, nLBD);
    cvt_f32_to_f16<<<(nLBD + 255) / 256, 256, 0, stream>>>(key_,  xk16, nLBD);
    cvt_f32_to_f16<<<(nLBD + 255) / 256, 256, 0, stream>>>(value, xv16, nLBD);
    cvt_f32_to_f16<<<(nDD  + 255) / 256, 256, 0, stream>>>(Wq, wq16, nDD);
    cvt_f32_to_f16<<<(nDD  + 255) / 256, 256, 0, stream>>>(Wk, wk16, nDD);
    cvt_f32_to_f16<<<(nDD  + 255) / 256, 256, 0, stream>>>(Wv, wv16, nDD);
    cvt_f32_to_f16<<<(nDD  + 255) / 256, 256, 0, stream>>>(Wo, wo16, nDD);

    // projections: (8192x1024)x(1024x1024)^T, head-split f16 outputs
    gemm_wmma<<<512, 512, 0, stream>>>(xq16, wq16, bq, qh, nullptr, 0.125f, 1); // *hd^-0.5
    gemm_wmma<<<512, 512, 0, stream>>>(xk16, wk16, bk, kh, nullptr, 1.0f, 1);
    gemm_wmma<<<512, 512, 0, stream>>>(xv16, wv16, bv, vh, nullptr, 1.0f, 1);

    // softmax stats
    stats_kernel<<<1024, 256, 0, stream>>>(qh, kh, mstat, lstat);

    // attention + avg_weights
    attn_kernel<<<dim3(128, 4), 512, 0, stream>>>(qh, kh, vh, mstat, lstat, at16, avg);

    // output projection -> f32 d_out
    gemm_wmma<<<512, 512, 0, stream>>>(at16, wo16, bo, nullptr, out, 1.0f, 0);
}
